// ClusterBinarySelection_21930103014175
// MI455X (gfx1250) — matrix operand order repaired and checked by reading, not compiled, not run on hardware
//
#include <hip/hip_runtime.h>
#include <hip/hip_bf16.h>
#include <math.h>

// ---------------- model constants ----------------
#define SQ   2048      // sequence length
#define DMC  768       // model dim
#define HC   12        // heads
#define DHC  64        // head dim
#define FFC  3072      // ffn dim
#define LC   4         // layers
#define WINC 256       // one-sided window
#define NCC  8         // S / WIN

// WMMA vector types (gfx1250, wave32)
typedef __attribute__((ext_vector_type(16))) __bf16 v16bf;
typedef __attribute__((ext_vector_type(8)))  float  v8f;

union FragAB { v16bf v; unsigned short u[16]; uint4 q[2]; };
union FragC  { v8f   v; float f[8]; };
union U4U    { uint4 q; unsigned short u[8]; };

static __device__ __forceinline__ unsigned short f2bf(float x) {
  unsigned int u = __float_as_uint(x);
  u += 0x7FFFu + ((u >> 16) & 1u);        // round-to-nearest-even
  return (unsigned short)(u >> 16);
}
static __device__ __forceinline__ float b2f(unsigned short u) {
  return __uint_as_float((unsigned int)u << 16);
}
static __device__ __forceinline__ float gelu_exact(float v) {
  return 0.5f * v * (1.0f + erff(v * 0.70710678118654752440f));
}

// ---------------- Tensor Data Mover support (guarded) ----------------
#if __has_builtin(__builtin_amdgcn_tensor_load_to_lds)
#define USE_TDM 1
#if __has_include(<hip/amd_detail/amd_gfx1250_TDM.h>)
#define TDM_ARGS6 1
#endif
typedef unsigned int u32x4_t __attribute__((ext_vector_type(4)));
typedef int          i32x8_t __attribute__((ext_vector_type(8)));
typedef int          i32x4_t __attribute__((ext_vector_type(4)));

// 2-D bf16 tile DMA: global (row-major, strideElems) -> LDS, with 16B pad
// after every 64B row chunk (so a 32-elem row lands at a 40-ushort stride).
static __device__ __forceinline__ void tdm_load_2d(unsigned lds_off, const void* gptr,
                                                   unsigned tile0, unsigned tile1,
                                                   unsigned strideElems) {
  unsigned long long ga = (unsigned long long)(uintptr_t)gptr;
  u32x4_t g0;
  g0.x = 1u;                                                  // count=1 (valid D#)
  g0.y = lds_off;                                             // LDS byte address
  g0.z = (unsigned)(ga & 0xFFFFFFFFu);                        // global_addr[31:0]
  g0.w = (unsigned)((ga >> 32) & 0x1FFFFFFu) | 0x80000000u;   // addr[56:32] | type=2
  const unsigned td0 = 0x40000000u, td1 = 0x40000000u;        // huge dims: no OOB clip
  i32x8_t g1;
  g1[0] = (int)(0x10000u | (1u << 20) | (3u << 22) | (3u << 25)); // 2B data|pad_en|int=16dw|amt=4dw
  g1[1] = (int)((td0 & 0xFFFFu) << 16);                       // tensor_dim0 lo16
  g1[2] = (int)((td0 >> 16) | ((td1 & 0xFFFFu) << 16));       // td0 hi16 | td1 lo16
  g1[3] = (int)((td1 >> 16) | (tile0 << 16));                 // td1 hi16 | tile_dim0
  g1[4] = (int)(tile1 & 0xFFFFu);                             // tile_dim1 (tile_dim2=0)
  g1[5] = (int)strideElems;                                   // tensor_dim0_stride lo32
  g1[6] = 0;
  g1[7] = 0;
  i32x4_t z4 = {0, 0, 0, 0};
#ifdef TDM_ARGS6
  i32x8_t z8 = {0, 0, 0, 0, 0, 0, 0, 0};
  __builtin_amdgcn_tensor_load_to_lds(g0, g1, z4, z4, z8, 0);
#else
  __builtin_amdgcn_tensor_load_to_lds(g0, g1, z4, z4, 0);
#endif
}
#if __has_builtin(__builtin_amdgcn_s_wait_tensorcnt)
#define TDM_WAIT2() __builtin_amdgcn_s_wait_tensorcnt(2)
#define TDM_WAIT0() __builtin_amdgcn_s_wait_tensorcnt(0)
#else
#define TDM_WAIT2() asm volatile("s_wait_tensorcnt 0x2" ::: "memory")
#define TDM_WAIT0() asm volatile("s_wait_tensorcnt 0x0" ::: "memory")
#endif
#else
#define USE_TDM 0
#endif

// ---------------- fp32 -> bf16 transpose (weights):  WT[n][k] = bf16(W[k][n]) --------
__global__ void convT_bf16(const float* __restrict__ W, unsigned short* __restrict__ WT,
                           int K, int N) {
  __shared__ float t[32][33];
  int n0 = blockIdx.x * 32, k0 = blockIdx.y * 32;
  for (int i = threadIdx.y; i < 32; i += 8)
    t[i][threadIdx.x] = W[(size_t)(k0 + i) * N + n0 + threadIdx.x];
  __syncthreads();
  for (int i = threadIdx.y; i < 32; i += 8)
    WT[(size_t)(n0 + i) * K + k0 + threadIdx.x] = f2bf(t[threadIdx.x][i]);
}

// ---------------- embedding + LayerNorm ----------------
__global__ __launch_bounds__(256) void embed_ln(const int* __restrict__ src,
                                                const float* __restrict__ ew,
                                                const float* __restrict__ ep,
                                                const float* __restrict__ gs,
                                                const float* __restrict__ gb,
                                                float* __restrict__ x,
                                                unsigned short* __restrict__ xb) {
  int s = blockIdx.x, tid = threadIdx.x;
  __shared__ float buf[DMC];
  __shared__ float red[256];
  __shared__ float s_mean, s_rstd;
  int tok = src[s];
  for (int d = tid; d < DMC; d += 256)
    buf[d] = ew[(size_t)tok * DMC + d] + ep[(size_t)(s + 2) * DMC + d];
  __syncthreads();
  float ps = 0.f, ps2 = 0.f;
  for (int d = tid; d < DMC; d += 256) { float v = buf[d]; ps += v; ps2 += v * v; }
  red[tid] = ps; __syncthreads();
  for (int o = 128; o > 0; o >>= 1) { if (tid < o) red[tid] += red[tid + o]; __syncthreads(); }
  if (tid == 0) s_mean = red[0] * (1.0f / DMC);
  __syncthreads();
  red[tid] = ps2; __syncthreads();
  for (int o = 128; o > 0; o >>= 1) { if (tid < o) red[tid] += red[tid + o]; __syncthreads(); }
  if (tid == 0) s_rstd = rsqrtf(red[0] * (1.0f / DMC) - s_mean * s_mean + 1e-5f);
  __syncthreads();
  float mean = s_mean, rstd = s_rstd;
  for (int d = tid; d < DMC; d += 256) {
    float v = (buf[d] - mean) * rstd * gs[d] + gb[d];
    x[(size_t)s * DMC + d] = v;
    xb[(size_t)s * DMC + d] = f2bf(v);
  }
}

// ---------------- residual add + LayerNorm ----------------
__global__ __launch_bounds__(256) void add_ln(const float* __restrict__ xin,
                                              const float* __restrict__ t,
                                              const float* __restrict__ gs,
                                              const float* __restrict__ gb,
                                              float* __restrict__ x,
                                              unsigned short* __restrict__ xb) {
  int s = blockIdx.x, tid = threadIdx.x;
  __shared__ float buf[DMC];
  __shared__ float red[256];
  __shared__ float s_mean, s_rstd;
  for (int d = tid; d < DMC; d += 256)
    buf[d] = xin[(size_t)s * DMC + d] + t[(size_t)s * DMC + d];
  __syncthreads();
  float ps = 0.f, ps2 = 0.f;
  for (int d = tid; d < DMC; d += 256) { float v = buf[d]; ps += v; ps2 += v * v; }
  red[tid] = ps; __syncthreads();
  for (int o = 128; o > 0; o >>= 1) { if (tid < o) red[tid] += red[tid + o]; __syncthreads(); }
  if (tid == 0) s_mean = red[0] * (1.0f / DMC);
  __syncthreads();
  red[tid] = ps2; __syncthreads();
  for (int o = 128; o > 0; o >>= 1) { if (tid < o) red[tid] += red[tid + o]; __syncthreads(); }
  if (tid == 0) s_rstd = rsqrtf(red[0] * (1.0f / DMC) - s_mean * s_mean + 1e-5f);
  __syncthreads();
  float mean = s_mean, rstd = s_rstd;
  for (int d = tid; d < DMC; d += 256) {
    float v = (buf[d] - mean) * rstd * gs[d] + gb[d];
    x[(size_t)s * DMC + d] = v;
    xb[(size_t)s * DMC + d] = f2bf(v);
  }
}

// ---------------- bf16 WMMA GEMM:  C[M,N] = A[M,K] @ W[K,N] + bias ----------------
// W passed TRANSPOSED (WT[n][k]) so both LDS tiles are row-contiguous.
// block tile 128x64, 8 waves of 32x32 (2x2 wmma tiles), K step 32.
// TDM path: double-buffered tensor_load_to_lds; fallback: cooperative b128 loads.
#define ASTR 40                 // padded LDS stride (ushorts) for 32-elem rows
#define ABUFU (128 * ASTR)      // A tile ushorts (5120)
#define BBUFU (64 * ASTR)       // B tile ushorts (2560)
#define GBUFU (ABUFU + BBUFU)   // one stage buffer (7680 ushorts = 15360 B)
#define GEMM_LDS_BYTES (2 * GBUFU * 2)

__global__ __launch_bounds__(256) void gemm_bf16(const unsigned short* __restrict__ A,
                                                 const unsigned short* __restrict__ Bt,
                                                 const float* __restrict__ bias,
                                                 float* __restrict__ Cf,
                                                 unsigned short* __restrict__ Cb,
                                                 int M, int N, int K, int act) {
  extern __shared__ unsigned short smg[];
  int tid = threadIdx.x;
  int w = tid >> 5, l = tid & 31;
  int bm = blockIdx.y * 128, bn = blockIdx.x * 64;
  int wm = (w & 3) * 32, wn = (w >> 2) * 32;

  FragC acc[2][2];
  #pragma unroll
  for (int i = 0; i < 2; i++)
    #pragma unroll
    for (int j = 0; j < 2; j++)
      #pragma unroll
      for (int r = 0; r < 8; r++) acc[i][j].f[r] = 0.f;

#if USE_TDM
  if (w == 0) {   // wave 0 drives the Tensor Data Mover (EXEC-independent DMA)
    tdm_load_2d(0u,            A  + (size_t)bm * K, 32u, 128u, (unsigned)K);
    tdm_load_2d(ABUFU * 2u,    Bt + (size_t)bn * K, 32u,  64u, (unsigned)K);
  }
#endif
  int ib = 0;
  for (int kt = 0; kt < K; kt += 32) {
#if USE_TDM
    __syncthreads();                       // alt buffer free for DMA overwrite
    if (w == 0) {
      if (kt + 32 < K) {
        unsigned nb = (unsigned)((ib ^ 1) * GBUFU * 2);
        tdm_load_2d(nb,             A  + (size_t)bm * K + kt + 32, 32u, 128u, (unsigned)K);
        tdm_load_2d(nb + ABUFU * 2u, Bt + (size_t)bn * K + kt + 32, 32u,  64u, (unsigned)K);
        TDM_WAIT2();                       // in-order TDM: current stage complete
      } else {
        TDM_WAIT0();
      }
    }
    __syncthreads();                       // current stage visible to all waves
#else
    {  // cooperative staging: A 128x32, B 64x32, both row-contiguous
      int arow = tid >> 1, aseg = tid & 1;
      const uint4* ap = (const uint4*)(A + (size_t)(bm + arow) * K + kt + aseg * 16);
      uint4* asd = (uint4*)(smg + arow * ASTR + aseg * 16);
      asd[0] = ap[0]; asd[1] = ap[1];
      int brow = tid >> 2, bseg = tid & 3;
      *(uint4*)(smg + ABUFU + brow * ASTR + bseg * 8) =
          *(const uint4*)(Bt + (size_t)(bn + brow) * K + kt + bseg * 8);
      if (kt + 32 < K) {
        __builtin_prefetch(A + (size_t)(bm + arow) * K + kt + 32 + aseg * 16, 0, 1);
        __builtin_prefetch(Bt + (size_t)(bn + brow) * K + kt + 32 + bseg * 8, 0, 1);
      }
      __syncthreads();
    }
#endif
    const unsigned short* As = smg + ib * GBUFU;
    const unsigned short* Bs = As + ABUFU;

    FragAB afr[2], bfr[2];
    int k0 = (l >> 4) * 8;
    #pragma unroll
    for (int ti = 0; ti < 2; ti++) {
      int lrow = wm + ti * 16 + (l & 15);
      afr[ti].q[0] = *(const uint4*)(As + lrow * ASTR + k0);
      afr[ti].q[1] = *(const uint4*)(As + lrow * ASTR + k0 + 16);
    }
    int ks = (l >> 4) * 8;
    #pragma unroll
    for (int tj = 0; tj < 2; tj++) {
      int brow = wn + tj * 16 + (l & 15);
      bfr[tj].q[0] = *(const uint4*)(Bs + brow * ASTR + ks);
      bfr[tj].q[1] = *(const uint4*)(Bs + brow * ASTR + ks + 16);
    }
    #pragma unroll
    for (int ti = 0; ti < 2; ti++)
      #pragma unroll
      for (int tj = 0; tj < 2; tj++)
        acc[ti][tj].v = __builtin_amdgcn_wmma_f32_16x16x32_bf16(
            false, afr[ti].v, false, bfr[tj].v, (short)0, acc[ti][tj].v, false, false);
#if USE_TDM
    ib ^= 1;
#else
    __syncthreads();
#endif
  }

  // epilogue: bias (+ GELU), fp32 and/or bf16 stores
  #pragma unroll
  for (int ti = 0; ti < 2; ti++) {
    #pragma unroll
    for (int tj = 0; tj < 2; tj++) {
      int col = bn + wn + tj * 16 + (l & 15);
      float bc = bias[col];
      #pragma unroll
      for (int r = 0; r < 8; r++) {
        int row = bm + wm + ti * 16 + ((l >> 4) * 8) + r;
        float v = acc[ti][tj].f[r] + bc;
        if (act) v = gelu_exact(v);
        if (Cf) Cf[(size_t)row * N + col] = v;
        if (Cb) Cb[(size_t)row * N + col] = f2bf(v);
      }
    }
  }
}

// ---------------- banded attention (WMMA QK^T + softmax + WMMA PV) ----------------
// grid: x = 32 (8 chunks x 4 query-slices of 64), y = 12 heads; 256 threads.
// LDS: scores 64x780 f32 (199680B) | V-band K-major 64x776 bf16 (99328B) | Q 64x72 bf16 (9216B)
#define SCSTR 780
#define VSTR  776
#define QSTR  72
#define SCBYTES (64 * SCSTR * 4)
#define VTBYTES (64 * VSTR * 2)
#define QSBYTES (64 * QSTR * 2)
#define ATTN_LDS_BYTES (SCBYTES + VTBYTES + QSBYTES)

__global__ __launch_bounds__(256) void attn_band(const unsigned short* __restrict__ Qb,
                                                 const unsigned short* __restrict__ Kb,
                                                 const unsigned short* __restrict__ Vb,
                                                 const int* __restrict__ mask,
                                                 unsigned short* __restrict__ ctxb) {
  extern __shared__ char smem[];
  float* sc = (float*)smem;
  unsigned short* vt = (unsigned short*)(smem + SCBYTES);
  unsigned short* qs = (unsigned short*)(smem + SCBYTES + VTBYTES);
  int tid = threadIdx.x, w = tid >> 5, l = tid & 31;
  int c = blockIdx.x >> 2, sub = blockIdx.x & 3, h = blockIdx.y;
  int qbase = c * WINC + sub * 64;
  int cbase = c * WINC - WINC;
  const float scale = 0.125f;   // 1/sqrt(64)

  // stage Q slice (64x64 bf16), vectorized
  for (int e = tid; e < 512; e += 256) {
    int r = e >> 3, seg = e & 7;
    *(uint4*)(qs + r * QSTR + seg * 8) =
        *(const uint4*)(Qb + (size_t)(qbase + r) * DMC + h * DHC + seg * 8);
  }
  // stage V band K-major: vt[d][k] = V[cbase+k][h*64+d]
  for (int e = tid; e < 768 * 8; e += 256) {
    int kr = e >> 3, seg = e & 7;
    int j = cbase + kr;
    int jc = j < 0 ? 0 : (j > SQ - 1 ? SQ - 1 : j);
    U4U vv;
    vv.q = *(const uint4*)(Vb + (size_t)jc * DMC + h * DHC + seg * 8);
    #pragma unroll
    for (int i = 0; i < 8; i++) vt[(seg * 8 + i) * VSTR + kr] = vv.u[i];
  }
  // global-key column (position 0)
  if (tid < 64) {
    int row = tid;
    float s = 0.f;
    for (int d = 0; d < DHC; d++)
      s += b2f(Qb[(size_t)(qbase + row) * DMC + h * DHC + d]) * b2f(Kb[h * DHC + d]);
    sc[row * SCSTR + 768] = (mask[0] != 0) ? s * scale : -1e9f;
  }
  __syncthreads();

  // Phase A: S = Q K^T over the 768-wide band; 4 M-tiles x 48 N-tiles
  for (int t = w; t < 4 * 48; t += 8) {
    int mt = t / 48, nt = t % 48, nb = nt * 16;
    FragC accs;
    #pragma unroll
    for (int r = 0; r < 8; r++) accs.f[r] = 0.f;
    int col = l & 15;
    int j = cbase + nb + col;
    int jc = j < 0 ? 0 : (j > SQ - 1 ? SQ - 1 : j);
    #pragma unroll
    for (int kk = 0; kk < 2; kk++) {
      FragAB a, b;
      int lrow = mt * 16 + (l & 15);
      int k0 = kk * 32 + ((l >> 4) * 8);
      a.q[0] = *(const uint4*)(qs + lrow * QSTR + k0);
      a.q[1] = *(const uint4*)(qs + lrow * QSTR + k0 + 16);
      const unsigned short* kp = Kb + (size_t)jc * DMC + h * DHC + kk * 32 + ((l >> 4) * 8);
      b.q[0] = *(const uint4*)kp;
      b.q[1] = *(const uint4*)(kp + 16);
      accs.v = __builtin_amdgcn_wmma_f32_16x16x32_bf16(
          false, a.v, false, b.v, (short)0, accs.v, false, false);
    }
    bool jok = (j >= 1) && (j < SQ) && (mask[j] != 0);   // pos 0 removed from band
    #pragma unroll
    for (int r = 0; r < 8; r++) {
      int lrow = mt * 16 + ((l >> 4) * 8) + r;
      int qpos = qbase + lrow;
      int dlt = j - qpos;
      bool ok = jok && (dlt <= WINC) && (dlt >= -WINC);
      sc[lrow * SCSTR + nb + col] = ok ? accs.f[r] * scale : -1e9f;
    }
  }
  __syncthreads();

  // Phase B: softmax over 769 entries per row (wave32 shuffles, hw v_exp)
  for (int rr = 0; rr < 8; rr++) {
    int row = w * 8 + rr;
    float m = -3.4e38f;
    for (int jx = l; jx <= 768; jx += 32) m = fmaxf(m, sc[row * SCSTR + jx]);
    #pragma unroll
    for (int o = 16; o > 0; o >>= 1) m = fmaxf(m, __shfl_xor(m, o));
    float ssum = 0.f;
    for (int jx = l; jx <= 768; jx += 32) {
      float e = __expf(sc[row * SCSTR + jx] - m);
      sc[row * SCSTR + jx] = e;
      ssum += e;
    }
    #pragma unroll
    for (int o = 16; o > 0; o >>= 1) ssum += __shfl_xor(ssum, o);
    float inv = 1.0f / ssum;
    for (int jx = l; jx <= 768; jx += 32) sc[row * SCSTR + jx] *= inv;
  }
  __syncthreads();

  // Phase C: O = P V (+ global column); 4 M-tiles x 4 Dh-tiles, 2 per wave
  for (int tt = 0; tt < 2; tt++) {
    int t = w * 2 + tt;
    int mt = t >> 2, nt = t & 3;
    FragC acco;
    #pragma unroll
    for (int r = 0; r < 8; r++) acco.f[r] = 0.f;
    int d = nt * 16 + (l & 15);
    for (int kk = 0; kk < 24; kk++) {
      FragAB a, b;
      int lrow = mt * 16 + (l & 15);
      int k0 = kk * 32 + ((l >> 4) * 8);
      #pragma unroll
      for (int i = 0; i < 8; i++) {
        a.u[i]     = f2bf(sc[lrow * SCSTR + k0 + i]);
        a.u[8 + i] = f2bf(sc[lrow * SCSTR + k0 + 16 + i]);
      }
      int ks = kk * 32 + ((l >> 4) * 8);
      const unsigned short* vp = vt + d * VSTR + ks;
      b.q[0] = *(const uint4*)vp;
      b.q[1] = *(const uint4*)(vp + 16);
      acco.v = __builtin_amdgcn_wmma_f32_16x16x32_bf16(
          false, a.v, false, b.v, (short)0, acco.v, false, false);
    }
    float v0 = b2f(Vb[h * DHC + d]);
    #pragma unroll
    for (int r = 0; r < 8; r++) {
      int lrow = mt * 16 + ((l >> 4) * 8) + r;
      int s = qbase + lrow;
      float o = acco.f[r] + sc[lrow * SCSTR + 768] * v0;
      ctxb[(size_t)s * DMC + h * DHC + d] = f2bf(o);
    }
  }
}

// ---------------- qg matvec:  out[d] = (x0 . W[:,d] + b[d]) * scale ----------------
__global__ __launch_bounds__(256) void matvec(const float* __restrict__ x0,
                                              const float* __restrict__ W,
                                              const float* __restrict__ b,
                                              float* __restrict__ out,
                                              int Kd, int Nd, float scl) {
  int d = blockIdx.x * 256 + threadIdx.x;
  if (d < Nd) {
    float a = 0.f;
    for (int k = 0; k < Kd; k++) a += x0[k] * W[(size_t)k * Nd + d];
    out[d] = (a + b[d]) * scl;
  }
}

// ---------------- global token attention (one block per head) ----------------
__global__ __launch_bounds__(256) void attn_global(const float* __restrict__ qg,
                                                   const unsigned short* __restrict__ Kg,
                                                   const unsigned short* __restrict__ Vg,
                                                   const int* __restrict__ mask,
                                                   unsigned short* __restrict__ ctxb) {
  int h = blockIdx.x, tid = threadIdx.x;
  __shared__ float gsc[SQ];
  __shared__ float red[256];
  __shared__ float s_m, s_inv;
  float lm = -3.4e38f;
  for (int s = tid; s < SQ; s += 256) {
    float a = 0.f;
    for (int d = 0; d < DHC; d++)
      a += qg[h * DHC + d] * b2f(Kg[(size_t)s * DMC + h * DHC + d]);
    float v = (mask[s] != 0) ? a : -1e9f;
    gsc[s] = v;
    lm = fmaxf(lm, v);
  }
  red[tid] = lm; __syncthreads();
  for (int o = 128; o > 0; o >>= 1) { if (tid < o) red[tid] = fmaxf(red[tid], red[tid + o]); __syncthreads(); }
  if (tid == 0) s_m = red[0];
  __syncthreads();
  float m = s_m, ls = 0.f;
  for (int s = tid; s < SQ; s += 256) {
    float e = __expf(gsc[s] - m);
    gsc[s] = e;
    ls += e;
  }
  red[tid] = ls; __syncthreads();
  for (int o = 128; o > 0; o >>= 1) { if (tid < o) red[tid] += red[tid + o]; __syncthreads(); }
  if (tid == 0) s_inv = 1.0f / red[0];
  __syncthreads();
  float inv = s_inv;
  if (tid < DHC) {
    int d = tid;
    float a = 0.f;
    for (int s = 0; s < SQ; s++)
      a += gsc[s] * b2f(Vg[(size_t)s * DMC + h * DHC + d]);
    ctxb[h * DHC + d] = f2bf(a * inv);   // row 0 of ctx
  }
}

// ---------------- classifier ----------------
__global__ __launch_bounds__(256) void cls_head(const float* __restrict__ x,
                                                const float* __restrict__ Wc,
                                                const float* __restrict__ bc,
                                                const int* __restrict__ mask,
                                                float* __restrict__ out) {
  int tid = threadIdx.x;
  __shared__ float red[256];
  float ps = 0.f;
  for (int d = tid; d < DMC; d += 256) ps += x[d] * Wc[d];
  red[tid] = ps; __syncthreads();
  for (int o = 128; o > 0; o >>= 1) { if (tid < o) red[tid] += red[tid + o]; __syncthreads(); }
  if (tid == 0) {
    float z = red[0] + bc[0];
    float sg = 1.0f / (1.0f + __expf(-z));
    float m0 = (float)mask[0];
    out[0] = sg * m0;
    out[1] = m0;
  }
}

// ---------------- host orchestration ----------------
static inline size_t aln(size_t v) { return (v + 255) & ~(size_t)255; }

extern "C" void kernel_launch(void* const* d_in, const int* in_sizes, int n_in,
                              void* d_out, int out_size, void* d_ws, size_t ws_size,
                              hipStream_t stream) {
  (void)in_sizes; (void)n_in; (void)out_size; (void)ws_size;
  const int*   src    = (const int*)  d_in[0];
  const int*   mask   = (const int*)  d_in[1];
  const float* ew     = (const float*)d_in[3];
  const float* ep     = (const float*)d_in[4];
  const float* lne_s  = (const float*)d_in[5];
  const float* lne_b  = (const float*)d_in[6];
  const float* Wq  = (const float*)d_in[7];  const float* bq  = (const float*)d_in[8];
  const float* Wk  = (const float*)d_in[9];  const float* bk  = (const float*)d_in[10];
  const float* Wv  = (const float*)d_in[11]; const float* bv  = (const float*)d_in[12];
  const float* Wo  = (const float*)d_in[13]; const float* bo  = (const float*)d_in[14];
  const float* Wqg = (const float*)d_in[15]; const float* bqg = (const float*)d_in[16];
  const float* Wkg = (const float*)d_in[17]; const float* bkg = (const float*)d_in[18];
  const float* Wvg = (const float*)d_in[19]; const float* bvg = (const float*)d_in[20];
  const float* ln1s = (const float*)d_in[21]; const float* ln1b = (const float*)d_in[22];
  const float* Wff1 = (const float*)d_in[23]; const float* bff1 = (const float*)d_in[24];
  const float* Wff2 = (const float*)d_in[25]; const float* bff2 = (const float*)d_in[26];
  const float* ln2s = (const float*)d_in[27]; const float* ln2b = (const float*)d_in[28];
  const float* Wcls = (const float*)d_in[29]; const float* bcls = (const float*)d_in[30];

  // workspace carve (~50 MB)
  char* p = (char*)d_ws;
  float* x            = (float*)p;           p += aln((size_t)SQ * DMC * 4);
  unsigned short* xb  = (unsigned short*)p;  p += aln((size_t)SQ * DMC * 2);
  unsigned short* qb  = (unsigned short*)p;  p += aln((size_t)SQ * DMC * 2);
  unsigned short* kb  = (unsigned short*)p;  p += aln((size_t)SQ * DMC * 2);
  unsigned short* vb  = (unsigned short*)p;  p += aln((size_t)SQ * DMC * 2);
  unsigned short* kgb = (unsigned short*)p;  p += aln((size_t)SQ * DMC * 2);
  unsigned short* vgb = (unsigned short*)p;  p += aln((size_t)SQ * DMC * 2);
  unsigned short* ctxb= (unsigned short*)p;  p += aln((size_t)SQ * DMC * 2);
  float* tmpD         = (float*)p;           p += aln((size_t)SQ * DMC * 4);
  unsigned short* h1b = (unsigned short*)p;  p += aln((size_t)SQ * FFC * 2);
  unsigned short* wbT = (unsigned short*)p;  p += aln((size_t)DMC * FFC * 2);
  float* qg           = (float*)p;           p += aln((size_t)DMC * 4);

  const dim3 gD(DMC / 64, SQ / 128);   // N=768 GEMMs
  const dim3 gF(FFC / 64, SQ / 128);   // N=3072 GEMM
  const dim3 tD(DMC / 32, DMC / 32);   // 768x768 weight transpose
  const dim3 t1(FFC / 32, DMC / 32);   // Wff1 (K=768,N=3072)
  const dim3 t2(DMC / 32, FFC / 32);   // Wff2 (K=3072,N=768)
  const dim3 tb(32, 8);

  embed_ln<<<SQ, 256, 0, stream>>>(src, ew, ep, lne_s, lne_b, x, xb);

  for (int l = 0; l < LC; l++) {
    const size_t oW = (size_t)l * DMC * DMC, oB = (size_t)l * DMC;
    const size_t oF1 = (size_t)l * DMC * FFC, oBF = (size_t)l * FFC;
    const size_t oF2 = (size_t)l * FFC * DMC;

    convT_bf16<<<tD, tb, 0, stream>>>(Wq + oW, wbT, DMC, DMC);
    gemm_bf16<<<gD, 256, GEMM_LDS_BYTES, stream>>>(xb, wbT, bq + oB, nullptr, qb, SQ, DMC, DMC, 0);
    convT_bf16<<<tD, tb, 0, stream>>>(Wk + oW, wbT, DMC, DMC);
    gemm_bf16<<<gD, 256, GEMM_LDS_BYTES, stream>>>(xb, wbT, bk + oB, nullptr, kb, SQ, DMC, DMC, 0);
    convT_bf16<<<tD, tb, 0, stream>>>(Wv + oW, wbT, DMC, DMC);
    gemm_bf16<<<gD, 256, GEMM_LDS_BYTES, stream>>>(xb, wbT, bv + oB, nullptr, vb, SQ, DMC, DMC, 0);
    convT_bf16<<<tD, tb, 0, stream>>>(Wkg + oW, wbT, DMC, DMC);
    gemm_bf16<<<gD, 256, GEMM_LDS_BYTES, stream>>>(xb, wbT, bkg + oB, nullptr, kgb, SQ, DMC, DMC, 0);
    convT_bf16<<<tD, tb, 0, stream>>>(Wvg + oW, wbT, DMC, DMC);
    gemm_bf16<<<gD, 256, GEMM_LDS_BYTES, stream>>>(xb, wbT, bvg + oB, nullptr, vgb, SQ, DMC, DMC, 0);

    attn_band<<<dim3(NCC * 4, HC), 256, ATTN_LDS_BYTES, stream>>>(qb, kb, vb, mask, ctxb);
    matvec<<<(DMC + 255) / 256, 256, 0, stream>>>(x, Wqg + oW, bqg + oB, qg, DMC, DMC, 0.125f);
    attn_global<<<HC, 256, 0, stream>>>(qg, kgb, vgb, mask, ctxb);

    convT_bf16<<<tD, tb, 0, stream>>>(Wo + oW, wbT, DMC, DMC);
    gemm_bf16<<<gD, 256, GEMM_LDS_BYTES, stream>>>(ctxb, wbT, bo + oB, tmpD, nullptr, SQ, DMC, DMC, 0);
    add_ln<<<SQ, 256, 0, stream>>>(x, tmpD, ln1s + oB, ln1b + oB, x, xb);

    convT_bf16<<<t1, tb, 0, stream>>>(Wff1 + oF1, wbT, DMC, FFC);
    gemm_bf16<<<gF, 256, GEMM_LDS_BYTES, stream>>>(xb, wbT, bff1 + oBF, nullptr, h1b, SQ, FFC, DMC, 1);
    convT_bf16<<<t2, tb, 0, stream>>>(Wff2 + oF2, wbT, FFC, DMC);
    gemm_bf16<<<gD, 256, GEMM_LDS_BYTES, stream>>>(h1b, wbT, bff2 + oB, tmpD, nullptr, SQ, DMC, FFC, 0);
    add_ln<<<SQ, 256, 0, stream>>>(x, tmpD, ln2s + oB, ln2b + oB, x, xb);
  }

  cls_head<<<1, 256, 0, stream>>>(x, Wcls, bcls, mask, (float*)d_out);
}
